// GRUModule_13709535608859
// MI455X (gfx1250) — compile-verified, matrix operand order checked by reading, not run on hardware
//
#include <hip/hip_runtime.h>

// GRU (single step, H=1) over N = 1,048,576 rows of 64 f32.
// Bandwidth-bound (256MB in @ 23.3TB/s ~= 11us floor); gi GEMM done with
// full-precision V_WMMA_F32_16X16X4_F32 (free at this intensity).

typedef float v2f  __attribute__((ext_vector_type(2)));
typedef float v4f  __attribute__((ext_vector_type(4)));
typedef float v8f  __attribute__((ext_vector_type(8)));
typedef _Float16 v16h __attribute__((ext_vector_type(16)));

#define THREADS        256
#define ROWS_PER_BLOCK 128            // 8 waves x 16 rows
#define LDS_STRIDE     68             // 64 + 4 pad: rows rotate 4 LDS banks

__global__ __launch_bounds__(THREADS)
void gru_wmma_kernel(const float* __restrict__ in,     // [N,64]: col0=h0, col1..63=x
                     const float* __restrict__ W_ih,   // [3,63]
                     const float* __restrict__ W_hh,   // [3]
                     const float* __restrict__ b_ih,   // [3]
                     const float* __restrict__ b_hh,   // [3]
                     const float* __restrict__ w_lin,  // [1]
                     const float* __restrict__ b_lin,  // [1]
                     float* __restrict__ out)          // [N]
{
    __shared__ float xs[ROWS_PER_BLOCK * LDS_STRIDE];  // 34.8 KB staged rows
    __shared__ float gates[ROWS_PER_BLOCK * 4];        // gi scatter buffer

    const int tid  = threadIdx.x;
    const int wave = tid >> 5;
    const int lane = tid & 31;
    const int half = lane >> 4;       // K-half selector (A) / row-half (D)
    const int nidx = lane & 15;       // A: row-in-tile; B/D: column
    const long row0 = (long)blockIdx.x * ROWS_PER_BLOCK;

    // ---- Stage 1: coalesced, non-temporal global -> LDS (128 rows x 256B) ----
    {
        const v4f* gsrc = (const v4f*)(in + row0 * 64);
        #pragma unroll
        for (int i = 0; i < 8; ++i) {
            int f  = tid + i * THREADS;            // 0..2047 float4s
            int r  = f >> 4;
            int c4 = f & 15;
            v4f v = __builtin_nontemporal_load(gsrc + r * 16 + c4);
            *(v4f*)(&xs[r * LDS_STRIDE + c4 * 4]) = v;
        }
    }
    __syncthreads();

    // ---- Stage 2: accumulator C = b_ih broadcast per output column ----
    v8f acc;
    {
        float cinit = (nidx < 3) ? b_ih[nidx] : 0.0f;
        #pragma unroll
        for (int j = 0; j < 8; ++j) acc[j] = cinit;
    }

    const float* arow = &xs[(wave * 16 + nidx) * LDS_STRIDE];

#if __has_builtin(__builtin_amdgcn_wmma_f32_16x16x4_f32)
    // ---- fp32 path: 16 x V_WMMA_F32_16X16X4_F32 over K=63 (chunk 16 padded) ----
    // B fragment (4x16 = W_ih^T slice): lane holds column nidx, K = 4c+2*half+{0,1}
    float bx[16], by[16];
    #pragma unroll
    for (int c = 0; c < 16; ++c) {
        int kg = 4 * c + 2 * half;
        float vx = 0.0f, vy = 0.0f;
        if (nidx < 3) {
            if (kg     < 63) vx = W_ih[nidx * 63 + kg];
            if (kg + 1 < 63) vy = W_ih[nidx * 63 + kg + 1];
        }
        bx[c] = vx; by[c] = vy;
    }
    #pragma unroll
    for (int c = 0; c < 16; ++c) {
        int kg = 4 * c + 2 * half;                 // x-index; memory col = 1+kg
        v2f a;
        a.x = arow[1 + kg];                        // kg   <= 62 always valid
        a.y = (kg + 1 < 63) ? arow[2 + kg] : 0.0f; // pad K=63 with 0
        v2f b; b.x = bx[c]; b.y = by[c];
        acc = __builtin_amdgcn_wmma_f32_16x16x4_f32(
                  false, a, false, b, (short)0, acc, false, false);
    }
#else
    // ---- fallback: 2 x V_WMMA_F32_16X16X32_F16 over padded K=64 ----
    #pragma unroll
    for (int c = 0; c < 2; ++c) {
        v16h a, b;
        #pragma unroll
        for (int i = 0; i < 16; ++i) {
            int kg = 32 * c + 8 * half + i + (i & 8); // elems 0-7: K=base+i; 8-15: +16
            a[i] = (kg < 63) ? (_Float16)arow[1 + kg] : (_Float16)0.0f;
            b[i] = (nidx < 3 && kg < 63) ? (_Float16)W_ih[nidx * 63 + kg]
                                         : (_Float16)0.0f;
        }
        acc = __builtin_amdgcn_wmma_f32_16x16x32_f16(
                  false, a, false, b, (short)0, acc, false, false);
    }
#endif

    // ---- Stage 4: scatter D (lane n = column, VGPR j = row j / j+8) to LDS ----
    if (nidx < 4) {
        #pragma unroll
        for (int j = 0; j < 8; ++j) {
            int m = wave * 16 + half * 8 + j;      // local row
            gates[m * 4 + nidx] = acc[j];
        }
    }
    __syncthreads();

    // ---- Stage 5: gate math + Linear(1,1); lanes 0-15 -> one row each ----
    if (half == 0) {
        int   m   = wave * 16 + nidx;
        float gi0 = gates[m * 4 + 0];
        float gi1 = gates[m * 4 + 1];
        float gi2 = gates[m * 4 + 2];
        float h0  = xs[m * LDS_STRIDE + 0];
        float gh0 = h0 * W_hh[0] + b_hh[0];
        float gh1 = h0 * W_hh[1] + b_hh[1];
        float gh2 = h0 * W_hh[2] + b_hh[2];
        float r   = 1.0f / (1.0f + __expf(-(gi0 + gh0)));
        float z   = 1.0f / (1.0f + __expf(-(gi1 + gh1)));
        float n   = tanhf(gi2 + r * gh2);
        float h1  = (1.0f - z) * n + z * h0;
        __builtin_nontemporal_store(h1 * w_lin[0] + b_lin[0], out + row0 + m);
    }
}

extern "C" void kernel_launch(void* const* d_in, const int* in_sizes, int n_in,
                              void* d_out, int out_size, void* d_ws, size_t ws_size,
                              hipStream_t stream) {
    const float* in    = (const float*)d_in[0];
    const float* W_ih  = (const float*)d_in[1];
    const float* W_hh  = (const float*)d_in[2];
    const float* b_ih  = (const float*)d_in[3];
    const float* b_hh  = (const float*)d_in[4];
    const float* w_lin = (const float*)d_in[5];
    const float* b_lin = (const float*)d_in[6];
    float* out = (float*)d_out;

    const int N = out_size;                         // 1,048,576 rows (divisible by 128)
    const int blocks = N / ROWS_PER_BLOCK;          // 8192
    gru_wmma_kernel<<<blocks, THREADS, 0, stream>>>(in, W_ih, W_hh, b_ih, b_hh,
                                                    w_lin, b_lin, out);
}
